// MultiHeadAttentionBlock_24618752540927
// MI455X (gfx1250) — matrix operand order repaired and checked by reading, not compiled, not run on hardware
//
#include <hip/hip_runtime.h>
#include <hip/hip_bf16.h>

// ---------------------------------------------------------------------------
// Linear attention block on gfx1250 (wave32, WMMA bf16->f32, double-buffered
// LDS-staged GEMMs).   B=4 S=8192 D=1024 H=16 DK=64
// ---------------------------------------------------------------------------

typedef __bf16 bf16;
typedef __attribute__((ext_vector_type(16))) __bf16 v16bf;
typedef __attribute__((ext_vector_type(8)))  __bf16 v8bf;
typedef __attribute__((ext_vector_type(8)))  float  v8f;

#define B_   4
#define S_   8192
#define D_   1024
#define H_   16
#define DK_  64
#define EPS_ 1e-6f

// ---- WMMA wrapper ----------------------------------------------------------
__device__ __forceinline__ v8f wmma_bf16(v16bf a, v16bf b, v8f c) {
  // (neg_a, A, neg_b, B, c_mod, C, reuse_a, reuse_b)
  return __builtin_amdgcn_wmma_f32_16x16x32_bf16(false, a, false, b,
                                                 (short)0, c, false, false);
}

// ---- fragment / staging helpers -------------------------------------------
__device__ __forceinline__ v16bf frag_from_bf16(const bf16* p0, const bf16* p1) {
  v8bf a = *(const v8bf*)p0;
  v8bf b = *(const v8bf*)p1;
  v16bf r;
#pragma unroll
  for (int i = 0; i < 8; ++i) { r[i] = a[i]; r[8 + i] = b[i]; }
  return r;
}

__device__ __forceinline__ v8bf cvt8(float4 a, float4 b) {
  v8bf r;
  r[0]=(bf16)a.x; r[1]=(bf16)a.y; r[2]=(bf16)a.z; r[3]=(bf16)a.w;
  r[4]=(bf16)b.x; r[5]=(bf16)b.y; r[6]=(bf16)b.z; r[7]=(bf16)b.w;
  return r;
}

// per-thread staging registers (16 A elements + 16 B elements)
struct StageRegs {
  float4 a0, a1, a2, a3;  // A as f32 (XF32 path)
  v8bf   h0, h1;          // A as bf16 (bf16 path)
  float4 b0, b1, b2, b3;  // B (weights) always f32
};

template <bool XF32>
__device__ __forceinline__ void load_stage(StageRegs& r, const void* Xv,
                                           const float* W, size_t xrow,
                                           size_t wrow, int k0, int sko) {
  const float* pW = W + wrow * D_ + k0 + sko;
  r.b0 = ((const float4*)pW)[0];
  r.b1 = ((const float4*)pW)[1];
  r.b2 = ((const float4*)pW)[2];
  r.b3 = ((const float4*)pW)[3];
  if constexpr (XF32) {
    const float* pX = (const float*)Xv + xrow * D_ + k0 + sko;
    r.a0 = ((const float4*)pX)[0];
    r.a1 = ((const float4*)pX)[1];
    r.a2 = ((const float4*)pX)[2];
    r.a3 = ((const float4*)pX)[3];
  } else {
    const bf16* pX = (const bf16*)Xv + xrow * D_ + k0 + sko;
    r.h0 = ((const v8bf*)pX)[0];
    r.h1 = ((const v8bf*)pX)[1];
  }
}

template <bool XF32>
__device__ __forceinline__ void store_stage(const StageRegs& r, bf16* dA,
                                            bf16* dB) {
  if constexpr (XF32) {
    ((v8bf*)dA)[0] = cvt8(r.a0, r.a1);
    ((v8bf*)dA)[1] = cvt8(r.a2, r.a3);
  } else {
    ((v8bf*)dA)[0] = r.h0;
    ((v8bf*)dA)[1] = r.h1;
  }
  ((v8bf*)dB)[0] = cvt8(r.b0, r.b1);
  ((v8bf*)dB)[1] = cvt8(r.b2, r.b3);
}

// ---------------------------------------------------------------------------
// Generic GEMM:  Y = f( X[M,1024] @ W[1024,1024]^T )
// Double-buffered LDS bf16 tiles, register prefetch, 1 barrier per k-step.
//   XF32  : X is f32 (else bf16)
//   RELU  : apply relu(x)+eps before store
//   SMODE : 0 = bf16 row-major [M,D]
//           1 = bf16 head-transposed [B,H,DK,S] (coalesced via LDS transpose)
//           2 = f32  row-major [M,D]            (final output)
// Block = 256 threads (8 waves); C tile 128x128; wave tile 32x64.
// ---------------------------------------------------------------------------
template <bool XF32, bool RELU, int SMODE>
__global__ __launch_bounds__(256) void gemm_xwT_kernel(
    const void* __restrict__ Xv, const float* __restrict__ W,
    void* __restrict__ Yv) {
  __shared__ bf16 lA[2][128 * 32];                    // [buf][row][k]
  __shared__ bf16 lB[2][128 * 32];                    // [buf][col][k]
  __shared__ bf16 tr[(SMODE == 1) ? 8 * 64 * 32 : 1]; // per-wave transpose buf

  const int tid  = threadIdx.x;
  const int lane = tid & 31;
  const int wid  = tid >> 5;
  const int lr   = lane & 15;
  const int hs   = lane >> 4;
  const int mw   = (wid & 3) * 32;   // wave row offset in block tile
  const int nw   = (wid >> 2) * 64;  // wave col offset in block tile
  const int m0   = blockIdx.x * 128 + mw;
  const int n0   = blockIdx.y * 128 + nw;

  // cooperative staging: 2 threads per row, 16 elements each
  const int    srow = tid >> 1;
  const int    sko  = (tid & 1) << 4;
  const size_t xrow = (size_t)(blockIdx.x * 128 + srow);
  const size_t wrow = (size_t)(blockIdx.y * 128 + srow);
  const int    soff = srow * 32 + sko;

  v8f acc[2][4] = {};
  StageRegs sr;

  // prologue: tile 0 -> buffer 0
  load_stage<XF32>(sr, Xv, W, xrow, wrow, 0, sko);
  store_stage<XF32>(sr, &lA[0][soff], &lB[0][soff]);
  __syncthreads();

  int p = 0;
  for (int k0 = 0; k0 < D_; k0 += 32) {
    const bool more = (k0 + 32) < D_;
    if (more) load_stage<XF32>(sr, Xv, W, xrow, wrow, k0 + 32, sko);

    v16bf af[2], bfr[4];
#pragma unroll
    for (int mt = 0; mt < 2; ++mt) {
      const bf16* pa = &lA[p][(mw + mt * 16 + lr) * 32 + hs * 8];
      af[mt] = frag_from_bf16(pa, pa + 16);
    }
#pragma unroll
    for (int nt = 0; nt < 4; ++nt) {
      const bf16* pb = &lB[p][(nw + nt * 16 + lr) * 32 + hs * 8];
      bfr[nt] = frag_from_bf16(pb, pb + 16);
    }
#pragma unroll
    for (int mt = 0; mt < 2; ++mt)
#pragma unroll
      for (int nt = 0; nt < 4; ++nt)
        acc[mt][nt] = wmma_bf16(af[mt], bfr[nt], acc[mt][nt]);

    if (more) {
      store_stage<XF32>(sr, &lA[1 - p][soff], &lB[1 - p][soff]);
      p ^= 1;
      __syncthreads();
    }
  }

  // C/D layout: VGPR r, lanes 0-15 -> M=r, N=lane; lanes 16-31 -> M=r+8.
  if constexpr (SMODE == 1) {
    // Transpose 32x64 wave tile through wave-private LDS, then store 64B
    // contiguous runs along s into Kt/Vt [B,H,DK,S].
    bf16* t = &tr[wid * 64 * 32];
#pragma unroll
    for (int mt = 0; mt < 2; ++mt)
#pragma unroll
      for (int nt = 0; nt < 4; ++nt)
#pragma unroll
        for (int r = 0; r < 8; ++r) {
          float val = acc[mt][nt][r];
          if constexpr (RELU) val = fmaxf(val, 0.f) + EPS_;
          const int rowi = mt * 16 + r + hs * 8;  // 0..31
          const int coli = nt * 16 + lr;          // 0..63
          t[coli * 32 + rowi] = (bf16)val;
        }
    asm volatile("s_wait_dscnt 0x0" ::: "memory");  // LDS writes -> reads
    const int b  = m0 >> 13;        // 32-aligned row block: same b, s run
    const int s0 = m0 & (S_ - 1);
#pragma unroll
    for (int cc = 0; cc < 2; ++cc) {
      const int c    = cc * 32 + lane;
      const int colg = n0 + c;
      const int h    = colg >> 6;
      const int d    = colg & (DK_ - 1);
      bf16* dst = (bf16*)Yv + (((size_t)(b * H_ + h) * DK_ + d) << 13) + s0;
#pragma unroll
      for (int j = 0; j < 4; ++j)
        *(v8bf*)(dst + j * 8) = *(const v8bf*)(t + c * 32 + j * 8);
    }
  } else {
#pragma unroll
    for (int mt = 0; mt < 2; ++mt)
#pragma unroll
      for (int nt = 0; nt < 4; ++nt)
#pragma unroll
        for (int r = 0; r < 8; ++r) {
          const int row = m0 + mt * 16 + r + hs * 8;
          const int col = n0 + nt * 16 + lr;
          float val = acc[mt][nt][r];
          if constexpr (RELU) val = fmaxf(val, 0.f) + EPS_;
          if constexpr (SMODE == 0)
            ((bf16*)Yv)[(size_t)row * D_ + col] = (bf16)val;
          else
            ((float*)Yv)[(size_t)row * D_ + col] = val;
        }
  }
}

// ---------------------------------------------------------------------------
// ksum[b,h] = sum over (d,s) of Kt[b,h,d,s].  Grid (B*H, 8); LDS tree reduce.
// ---------------------------------------------------------------------------
__global__ __launch_bounds__(256) void ksum_kernel(const bf16* __restrict__ Kt,
                                                   float* __restrict__ ksum) {
  __shared__ float red[256];
  const int bh = blockIdx.x;
  const size_t chunk = (size_t)DK_ * S_ / 8;  // 65536 elements
  const bf16* p = Kt + (size_t)bh * DK_ * S_ + (size_t)blockIdx.y * chunk;
  float s = 0.f;
  for (size_t i = (size_t)threadIdx.x * 8; i < chunk; i += 256 * 8) {
    v8bf vv = *(const v8bf*)(p + i);
#pragma unroll
    for (int j = 0; j < 8; ++j) s += (float)vv[j];
  }
  red[threadIdx.x] = s;
  __syncthreads();
  for (int off = 128; off; off >>= 1) {
    if ((int)threadIdx.x < off) red[threadIdx.x] += red[threadIdx.x + off];
    __syncthreads();
  }
  if (threadIdx.x == 0) atomicAdd(&ksum[bh], red[0]);
}

// ---------------------------------------------------------------------------
// kv[b,h,d,e] = sum_n Kt[b,h,d,n] * Vt[b,h,e,n]
// Grid (B*H, 4); block 256 = 8 waves = 4 e-tiles x 2 n-halves; f32 atomics.
// n-loop unrolled x2 so two iterations of fragments are in flight.
// ---------------------------------------------------------------------------
__global__ __launch_bounds__(256) void kv_head_kernel(
    const bf16* __restrict__ Kt, const bf16* __restrict__ Vt,
    float* __restrict__ kv) {
  const int lane = threadIdx.x & 31;
  const int wid  = threadIdx.x >> 5;
  const int lr   = lane & 15;
  const int hs   = lane >> 4;
  const int bh   = blockIdx.x;
  const int et   = wid & 3;   // e (N) tile
  const int nsub = wid >> 2;  // n half of this block's chunk
  const size_t base = (size_t)bh * DK_ * S_;
  const int n0 = blockIdx.y * 2048 + nsub * 1024;

  v8f acc[4] = {};
  for (int n = n0; n < n0 + 1024; n += 64) {
    v16bf a0[4], a1[4];
#pragma unroll
    for (int mt = 0; mt < 4; ++mt) {
      const bf16* p = Kt + base + (size_t)(mt * 16 + lr) * S_ + n + hs * 8;
      a0[mt] = frag_from_bf16(p, p + 16);
      a1[mt] = frag_from_bf16(p + 32, p + 48);
    }
    const bf16* pb = Vt + base + (size_t)(et * 16 + lr) * S_ + n + hs * 8;
    v16bf b0 = frag_from_bf16(pb, pb + 16);
    v16bf b1 = frag_from_bf16(pb + 32, pb + 48);
#pragma unroll
    for (int mt = 0; mt < 4; ++mt) acc[mt] = wmma_bf16(a0[mt], b0, acc[mt]);
#pragma unroll
    for (int mt = 0; mt < 4; ++mt) acc[mt] = wmma_bf16(a1[mt], b1, acc[mt]);
  }

  float* kvh = kv + (size_t)bh * DK_ * DK_;
#pragma unroll
  for (int mt = 0; mt < 4; ++mt)
#pragma unroll
    for (int r = 0; r < 8; ++r) {
      const int d = mt * 16 + r + hs * 8;
      const int e = et * 16 + lr;
      atomicAdd(&kvh[d * DK_ + e], acc[mt][r]);
    }
}

// ---------------------------------------------------------------------------
// ctx[b,s,h*64+e] = (Q[b,s,h,:] @ kv[b,h]) / (rowsum(Q)*ksum[b,h] + eps)
// Grid (B*H, S/128); kv staged once per block into LDS (transposed, bf16).
// ---------------------------------------------------------------------------
__global__ __launch_bounds__(256) void qkv_norm_kernel(
    const bf16* __restrict__ Qb, const float* __restrict__ kv,
    const float* __restrict__ ksum, bf16* __restrict__ ctx) {
  __shared__ bf16 kvT[DK_ * DK_];  // [e][d]
  const int tid  = threadIdx.x;
  const int lane = tid & 31;
  const int wid  = tid >> 5;
  const int lr   = lane & 15;
  const int hs   = lane >> 4;
  const int bh   = blockIdx.x;
  const int b    = bh >> 4;
  const int h    = bh & 15;
  const size_t g0 = (size_t)b * S_ + blockIdx.y * 128 + wid * 16;
  const float* kvh = kv + (size_t)bh * DK_ * DK_;

  {  // stage kv -> LDS transposed bf16 (16 elements per thread)
    const int d  = tid >> 2;
    const int e0 = (tid & 3) * 16;
    const float* src = kvh + d * DK_ + e0;
    float tmp[16];
    *(float4*)&tmp[0]  = ((const float4*)src)[0];
    *(float4*)&tmp[4]  = ((const float4*)src)[1];
    *(float4*)&tmp[8]  = ((const float4*)src)[2];
    *(float4*)&tmp[12] = ((const float4*)src)[3];
#pragma unroll
    for (int j = 0; j < 16; ++j) kvT[(e0 + j) * DK_ + d] = (bf16)tmp[j];
  }
  __syncthreads();

  // A fragments over K=64 (two k-steps) + per-row sums
  v16bf a[2];
  float part = 0.f;
#pragma unroll
  for (int ks = 0; ks < 2; ++ks) {
    const bf16* p = Qb + (g0 + lr) * D_ + h * DK_ + ks * 32 + hs * 8;
    a[ks] = frag_from_bf16(p, p + 16);
#pragma unroll
    for (int i = 0; i < 16; ++i) part += (float)a[ks][i];
  }
  const float rowsum = part + __shfl_xor(part, 16, 32);
  const float nrm = 1.f / (rowsum * ksum[bh] + EPS_);

  v8f acc[4] = {};
#pragma unroll
  for (int ks = 0; ks < 2; ++ks) {
#pragma unroll
    for (int nt = 0; nt < 4; ++nt) {
      const bf16* p = &kvT[(nt * 16 + lr) * DK_ + ks * 32 + hs * 8];
      acc[nt] = wmma_bf16(a[ks], frag_from_bf16(p, p + 16), acc[nt]);
    }
  }

  // normalizer for C row m=r+hs*8 lives in lane (r+hs*8)
  float nr[8];
#pragma unroll
  for (int r = 0; r < 8; ++r) nr[r] = __shfl(nrm, r + hs * 8, 32);

#pragma unroll
  for (int nt = 0; nt < 4; ++nt)
#pragma unroll
    for (int r = 0; r < 8; ++r) {
      const size_t g = g0 + r + hs * 8;
      ctx[g * D_ + h * DK_ + nt * 16 + lr] = (bf16)(acc[nt][r] * nr[r]);
    }
}

// ---------------------------------------------------------------------------
extern "C" void kernel_launch(void* const* d_in, const int* in_sizes, int n_in,
                              void* d_out, int out_size, void* d_ws,
                              size_t ws_size, hipStream_t stream) {
  (void)in_sizes; (void)n_in; (void)out_size; (void)ws_size;

  const float* q  = (const float*)d_in[0];
  const float* k  = (const float*)d_in[1];
  const float* v  = (const float*)d_in[2];
  const float* Wq = (const float*)d_in[3];
  const float* Wk = (const float*)d_in[4];
  const float* Wv = (const float*)d_in[5];
  const float* Wo = (const float*)d_in[6];

  char* ws = (char*)d_ws;
  const size_t elems = (size_t)B_ * S_ * D_;  // 33,554,432
  bf16*  Qb  = (bf16*)(ws);                   // [M,D] row-major
  bf16*  Kt  = (bf16*)(ws + elems * 2);       // [B,H,DK,S]
  bf16*  Vt  = (bf16*)(ws + elems * 4);       // [B,H,DK,S]
  bf16*  ctx = (bf16*)(ws + elems * 6);       // [M,D] row-major
  float* kvb = (float*)(ws + elems * 8);      // [B,H,DK,DK]
  float* ksm = (float*)(ws + elems * 8 + (size_t)B_ * H_ * DK_ * DK_ * 4);

  // zero atomic accumulators (kv + ksum are contiguous)
  hipMemsetAsync(kvb, 0,
                 (size_t)B_ * H_ * DK_ * DK_ * 4 + (size_t)B_ * H_ * 4, stream);

  const dim3 gg(B_ * S_ / 128, D_ / 128), bb(256);
  gemm_xwT_kernel<true,  true,  0><<<gg, bb, 0, stream>>>(q, Wq, Qb);
  gemm_xwT_kernel<true,  true,  1><<<gg, bb, 0, stream>>>(k, Wk, Kt);
  gemm_xwT_kernel<true,  false, 1><<<gg, bb, 0, stream>>>(v, Wv, Vt);
  ksum_kernel<<<dim3(B_ * H_, 8), bb, 0, stream>>>(Kt, ksm);
  kv_head_kernel<<<dim3(B_ * H_, 4), bb, 0, stream>>>(Kt, Vt, kvb);
  qkv_norm_kernel<<<dim3(B_ * H_, S_ / 128), bb, 0, stream>>>(Qb, kvb, ksm, ctx);
  gemm_xwT_kernel<false, false, 2><<<gg, bb, 0, stream>>>(ctx, Wo, (float*)d_out);
}
